// Timeline_37477884625304
// MI455X (gfx1250) — matrix operand order repaired and checked by reading, not compiled, not run on hardware
//
#include <hip/hip_runtime.h>
#include <hip/hip_bf16.h>
#include <math.h>

// ---------------------------------------------------------------------------
// Problem dims (from reference)
// ---------------------------------------------------------------------------
constexpr int Bn = 32, Vn = 128, Cn = 64, En = 256, Hn = 256, An = 128;
constexpr int VOCAB1 = 20001;          // VOCAB + 1 rows in emb/decay/initial
constexpr float BIGF = 1e20f;

typedef __attribute__((ext_vector_type(16))) __bf16 bf16x16;
typedef __attribute__((ext_vector_type(8)))  __bf16 bf16x8;
typedef __attribute__((ext_vector_type(8)))  float  f32x8;

#define DEV static __device__ __forceinline__

// ---------------------------------------------------------------------------
// WMMA fragment helpers (CDNA5 v_wmma_f32_16x16x32_bf16, wave32)
// A (16x32, 16-bit): lane m<16 holds row m, K = kb+0..7 (v[0..7]) and
//   kb+16..23 (v[8..15]); lanes 16..31 hold K = kb+8..15 / kb+24..31.
// B (32x16, 16-bit): lane l holds column n=l&15; same K split as A:
//   lanes 0-15 K=kb+0..7 / kb+16..23, lanes 16-31 K=kb+8..15 / kb+24..31.
// C/D (16x16 f32): lane l<16 -> col l, rows 0..7; lane l+16 -> col l, rows 8..15.
// ---------------------------------------------------------------------------
DEV bf16x16 frag_a16(const __bf16* base, int ld, int mbase, int kbase, int lane) {
  const int m = lane & 15, half = lane >> 4;
  const __bf16* p0 = base + (size_t)(mbase + m) * ld + kbase + half * 8;
  bf16x8 x0 = *(const bf16x8*)p0;
  bf16x8 x1 = *(const bf16x8*)(p0 + 16);
  bf16x16 r;
#pragma unroll
  for (int i = 0; i < 8; ++i) { r[i] = x0[i]; r[i + 8] = x1[i]; }
  return r;
}

// src is row-major [N][K] (row n == column n of the logical B matrix)
DEV bf16x16 frag_b16(const __bf16* base, int ld, int nbase, int kbase, int lane) {
  const int n = lane & 15, half = lane >> 4;
  const __bf16* p = base + (size_t)(nbase + n) * ld + kbase + half * 8;
  bf16x8 x0 = *(const bf16x8*)p;
  bf16x8 x1 = *(const bf16x8*)(p + 16);
  bf16x16 r;
#pragma unroll
  for (int i = 0; i < 8; ++i) { r[i] = x0[i]; r[i + 8] = x1[i]; }
  return r;
}

DEV f32x8 wmma_bf16(bf16x16 a, bf16x16 b, f32x8 c) {
  return __builtin_amdgcn_wmma_f32_16x16x32_bf16(
      /*neg_a=*/false, a, /*neg_b=*/false, b,
      /*c_mod=*/(short)0, c, /*reuse_a=*/false, /*reuse_b=*/false);
}

DEV float sigmoidf(float x) { return 1.0f / (1.0f + expf(-x)); }

// ---------------------------------------------------------------------------
// Kernel 0: fp32 -> bf16 conversion (weights / embedding table)
// ---------------------------------------------------------------------------
__global__ __launch_bounds__(256) void cvt_bf16_kernel(const float* __restrict__ src,
                                                       __bf16* __restrict__ dst, int n) {
  for (int i = blockIdx.x * 256 + threadIdx.x; i < n; i += gridDim.x * 256)
    dst[i] = (__bf16)src[i];
}

// ---------------------------------------------------------------------------
// Kernel 1: per-(b,v) code attention -> vv[b*V+v][E] (bf16)
// One 256-thread (8-wave) workgroup per visit. All GEMMs on WMMA bf16.
// ---------------------------------------------------------------------------
struct Phase2 {
  float  scores[Cn][Cn];   // 16 KB
  __bf16 attn[Cn][Cn];     //  8 KB
};
union AttnU {
  __bf16 te[Cn][En];       // 32 KB  (dead after Q/K GEMMs -> reused by Phase2)
  Phase2 p2;
};

__global__ __launch_bounds__(256) void attn_kernel(
    const int* __restrict__ codes, const float* __restrict__ code_mask,
    const float* __restrict__ time_vals, const float* __restrict__ decay,
    const float* __restrict__ initial, const __bf16* __restrict__ emb_bf,
    const __bf16* __restrict__ wq_bf, const __bf16* __restrict__ wk_bf,
    __bf16* __restrict__ vv_bf) {
  __shared__ AttnU u;                       // 32 KB
  __shared__ __bf16 teT[En][Cn];            // 32 KB (E-major transpose of te)
  __shared__ __bf16 Qs[Cn][An];             // 16 KB
  __shared__ __bf16 Ks[Cn][An];             // 16 KB
  __shared__ float  temp2[Cn];              // sigmoid(decay*t+ini) * gate
  __shared__ float  cmsh[Cn];
  __shared__ float  vvs[En];
  __shared__ int    codes_sh[Cn];

  const int bv = blockIdx.x;                // b*Vn + v
  const int tid = threadIdx.x;
  const int wave = tid >> 5, lane = tid & 31;

  if (tid < Cn) {
    codes_sh[tid] = codes[(size_t)bv * Cn + tid];
    cmsh[tid] = code_mask[(size_t)bv * Cn + tid];
  }
  __syncthreads();

  // ---- gather embedding rows (bf16) into te and teT; temp2; zero vv ----
  for (int idx = tid; idx < Cn * (En / 8); idx += 256) {
    const int c = idx / (En / 8);
    const int e8 = (idx % (En / 8)) * 8;
    bf16x8 row = *(const bf16x8*)(emb_bf + (size_t)codes_sh[c] * En + e8);
    *(bf16x8*)&u.te[c][e8] = row;
#pragma unroll
    for (int j = 0; j < 8; ++j) teT[e8 + j][c] = row[j];
  }
  if (tid < Cn) {
    const float t = time_vals[bv];
    const int code = codes_sh[tid];
    const float gate = (cmsh[tid] - BIGF) / (-BIGF);   // 1 for real codes
    temp2[tid] = gate * sigmoidf(decay[code] * t + initial[code]);
  }
  if (tid < En) vvs[tid] = 0.0f;
  __syncthreads();

  // ---- Q = te @ WQ^T, K = te @ WK^T   ([64,256] x [256,128]) ----
  for (int t = wave; t < 2 * 32; t += 8) {
    const int which = t >> 5;                 // 0: Q, 1: K
    const int mt = (t & 31) >> 3, nt = t & 7; // 4 x 8 tiles
    const __bf16* W = which ? wk_bf : wq_bf;
    f32x8 acc = {};
#pragma unroll
    for (int k = 0; k < En; k += 32) {
      bf16x16 a = frag_a16(&u.te[0][0], En, mt * 16, k, lane);
      bf16x16 b = frag_b16(W, En, nt * 16, k, lane);
      acc = wmma_bf16(a, b, acc);
    }
    const int col = nt * 16 + (lane & 15);
    const int rowb = mt * 16 + (lane >> 4) * 8;
    __bf16(*dst)[An] = which ? Ks : Qs;
#pragma unroll
    for (int r = 0; r < 8; ++r) dst[rowb + r][col] = (__bf16)acc[r];
  }
  __syncthreads();

  // ---- scores = Q @ K^T  ([64,128] x [128,64]); te is dead now ----
  for (int t = wave; t < 16; t += 8) {
    const int mt = t >> 2, nt = t & 3;
    f32x8 acc = {};
#pragma unroll
    for (int k = 0; k < An; k += 32) {
      bf16x16 a = frag_a16(&Qs[0][0], An, mt * 16, k, lane);
      bf16x16 b = frag_b16(&Ks[0][0], An, nt * 16, k, lane);  // row d of K = col d of K^T
      acc = wmma_bf16(a, b, acc);
    }
    const int col = nt * 16 + (lane & 15);
    const int rowb = mt * 16 + (lane >> 4) * 8;
#pragma unroll
    for (int r = 0; r < 8; ++r) u.p2.scores[rowb + r][col] = acc[r];
  }
  __syncthreads();

  // ---- row softmax: val = (s - cm[row] - cm[col]) / sqrt(A) ----
  if (tid < Cn) {
    const float inv = rsqrtf((float)An);
    const float cmr = cmsh[tid];
    float mx = -3.4e38f;
#pragma unroll 4
    for (int d = 0; d < Cn; ++d) {
      float s = (u.p2.scores[tid][d] - cmr - cmsh[d]) * inv;
      u.p2.scores[tid][d] = s;
      mx = fmaxf(mx, s);
    }
    float sum = 0.0f;
#pragma unroll 4
    for (int d = 0; d < Cn; ++d) {
      float e = expf(u.p2.scores[tid][d] - mx);
      u.p2.scores[tid][d] = e;
      sum += e;
    }
    const float rs = 1.0f / sum;
#pragma unroll 4
    for (int d = 0; d < Cn; ++d) u.p2.attn[tid][d] = (__bf16)(u.p2.scores[tid][d] * rs);
  }
  __syncthreads();

  // ---- fe = attn @ te, fused with vv[e] += temp2[c] * fe[c][e] ----
  for (int t = wave; t < 64; t += 8) {
    const int mt = t >> 4, nt = t & 15;       // 4 x 16 tiles over [64, 256]
    f32x8 acc = {};
#pragma unroll
    for (int k = 0; k < Cn; k += 32) {
      bf16x16 a = frag_a16(&u.p2.attn[0][0], Cn, mt * 16, k, lane);
      bf16x16 b = frag_b16(&teT[0][0], Cn, nt * 16, k, lane);  // teT row e = col e of te
      acc = wmma_bf16(a, b, acc);
    }
    const int rowb = mt * 16 + (lane >> 4) * 8;
    float s = 0.0f;
#pragma unroll
    for (int r = 0; r < 8; ++r) s += acc[r] * temp2[rowb + r];
    atomicAdd(&vvs[nt * 16 + (lane & 15)], s);   // ds_add_f32
  }
  __syncthreads();

  vv_bf[(size_t)bv * En + tid] = (__bf16)vvs[tid];
}

// ---------------------------------------------------------------------------
// Kernel 2: xg[dir][row=b*V+v][4H] = vv @ Wih^T + bias   (WMMA)
// grid = (M/16 = 256, 2 dirs), 8 waves x 8 N-tiles each.
// ---------------------------------------------------------------------------
__global__ __launch_bounds__(256) void xg_kernel(
    const __bf16* __restrict__ vv_bf, const __bf16* __restrict__ wih_f,
    const __bf16* __restrict__ wih_b, const float* __restrict__ b_f,
    const float* __restrict__ b_b, float* __restrict__ xg) {
  const int dir = blockIdx.y;
  const __bf16* W = dir ? wih_b : wih_f;
  const float* bias = dir ? b_b : b_f;
  float* dst = xg + (size_t)dir * (Bn * Vn) * (4 * Hn);
  const int mt = blockIdx.x;                  // rows 16*mt .. 16*mt+15 of 4096
  const int wave = threadIdx.x >> 5, lane = threadIdx.x & 31;

  for (int nt = wave; nt < (4 * Hn) / 16; nt += 8) {
    f32x8 acc = {};
#pragma unroll
    for (int k = 0; k < En; k += 32) {
      bf16x16 a = frag_a16(vv_bf, En, mt * 16, k, lane);
      bf16x16 b = frag_b16(W, En, nt * 16, k, lane);
      acc = wmma_bf16(a, b, acc);
    }
    const int col = nt * 16 + (lane & 15);
    const int rowb = mt * 16 + (lane >> 4) * 8;
    const float bv = bias[col];
#pragma unroll
    for (int r = 0; r < 8; ++r)
      dst[(size_t)(rowb + r) * (4 * Hn) + col] = acc[r] + bv;
  }
}

// ---------------------------------------------------------------------------
// Kernel 3: bidirectional LSTM scan + masked sum-pool.
// grid = 2 (dir), 256 threads. Recurrent GEMM h@Whh^T via WMMA each step;
// c-state and output accumulator live in registers (each thread owns 32
// (b,h) cells); h (bf16) and gate pre-activations staged in LDS.
// ---------------------------------------------------------------------------
__global__ __launch_bounds__(256) void lstm_kernel(
    const __bf16* __restrict__ whh_f, const __bf16* __restrict__ whh_b,
    const float* __restrict__ xg, const float* __restrict__ visit_mask,
    float* __restrict__ out) {
  __shared__ float  gates[Bn][4 * Hn];   // 128 KB
  __shared__ __bf16 hs[Bn][Hn];          //  16 KB

  const int dir = blockIdx.x;
  const __bf16* W = dir ? whh_b : whh_f;
  const float* xgd = xg + (size_t)dir * (Bn * Vn) * (4 * Hn);
  const int tid = threadIdx.x, wave = tid >> 5, lane = tid & 31;

  float cpriv[32], hsum[32];
#pragma unroll
  for (int j = 0; j < 32; ++j) { cpriv[j] = 0.0f; hsum[j] = 0.0f; }
  for (int i = tid; i < Bn * Hn; i += 256) (&hs[0][0])[i] = (__bf16)0.0f;
  __syncthreads();

  for (int step = 0; step < Vn; ++step) {
    const int v = dir ? (Vn - 1 - step) : step;

    // gates = hs @ Whh^T  ([32,256] x [256,1024]) : 2 x 64 tiles
    for (int t = wave; t < 128; t += 8) {
      const int mt = t >> 6, nt = t & 63;
      f32x8 acc = {};
#pragma unroll
      for (int k = 0; k < Hn; k += 32) {
        bf16x16 a = frag_a16(&hs[0][0], Hn, mt * 16, k, lane);
        bf16x16 b = frag_b16(W, Hn, nt * 16, k, lane);
        acc = wmma_bf16(a, b, acc);
      }
      const int col = nt * 16 + (lane & 15);
      const int rowb = mt * 16 + (lane >> 4) * 8;
#pragma unroll
      for (int r = 0; r < 8; ++r) gates[rowb + r][col] = acc[r];
    }
    __syncthreads();

    // elementwise LSTM cell; thread owns flat cells tid + 256*j
#pragma unroll
    for (int j = 0; j < 32; ++j) {
      const int flat = tid + 256 * j;
      const int bb = flat >> 8;       // batch
      const int hh = flat & 255;      // hidden index
      const float* xrow = xgd + ((size_t)bb * Vn + v) * (4 * Hn);
      const float gi = gates[bb][hh]            + xrow[hh];
      const float gf = gates[bb][Hn + hh]       + xrow[Hn + hh];
      const float gg = gates[bb][2 * Hn + hh]   + xrow[2 * Hn + hh];
      const float go = gates[bb][3 * Hn + hh]   + xrow[3 * Hn + hh];
      const float c = sigmoidf(gf) * cpriv[j] + sigmoidf(gi) * tanhf(gg);
      cpriv[j] = c;
      const float h = sigmoidf(go) * tanhf(c);
      hsum[j] += visit_mask[bb * Vn + v] * h;
      hs[bb][hh] = (__bf16)h;
    }
    __syncthreads();
  }

#pragma unroll
  for (int j = 0; j < 32; ++j) {
    const int flat = tid + 256 * j;
    const int bb = flat >> 8, hh = flat & 255;
    out[(size_t)bb * (2 * Hn) + dir * Hn + hh] = hsum[j];
  }
}

// ---------------------------------------------------------------------------
// Host launcher
// ---------------------------------------------------------------------------
extern "C" void kernel_launch(void* const* d_in, const int* in_sizes, int n_in,
                              void* d_out, int out_size, void* d_ws, size_t ws_size,
                              hipStream_t stream) {
  const int*   codes      = (const int*)d_in[0];
  const float* code_mask  = (const float*)d_in[1];
  const float* time_vals  = (const float*)d_in[2];
  const float* visit_mask = (const float*)d_in[3];
  const float* emb        = (const float*)d_in[4];
  const float* WQ         = (const float*)d_in[5];
  const float* WK         = (const float*)d_in[6];
  const float* decay      = (const float*)d_in[7];
  const float* initial    = (const float*)d_in[8];
  const float* Wih_f      = (const float*)d_in[9];
  const float* Whh_f      = (const float*)d_in[10];
  const float* b_f        = (const float*)d_in[11];
  const float* Wih_b      = (const float*)d_in[12];
  const float* Whh_b      = (const float*)d_in[13];
  const float* b_b        = (const float*)d_in[14];
  float* out = (float*)d_out;

  char* ws = (char*)d_ws;
  size_t off = 0;
  auto take = [&](size_t bytes) {
    size_t o = off;
    off += (bytes + 255) & ~(size_t)255;
    return o;
  };
  __bf16* emb_bf   = (__bf16*)(ws + take((size_t)VOCAB1 * En * 2));
  __bf16* wq_bf    = (__bf16*)(ws + take((size_t)An * En * 2));
  __bf16* wk_bf    = (__bf16*)(ws + take((size_t)An * En * 2));
  __bf16* wih_f_bf = (__bf16*)(ws + take((size_t)4 * Hn * En * 2));
  __bf16* whh_f_bf = (__bf16*)(ws + take((size_t)4 * Hn * Hn * 2));
  __bf16* wih_b_bf = (__bf16*)(ws + take((size_t)4 * Hn * En * 2));
  __bf16* whh_b_bf = (__bf16*)(ws + take((size_t)4 * Hn * Hn * 2));
  __bf16* vv_bf    = (__bf16*)(ws + take((size_t)Bn * Vn * En * 2));
  float*  xg       = (float*)(ws + take((size_t)2 * Bn * Vn * 4 * Hn * 4));

  auto cvt = [&](const float* s, __bf16* d, int n) {
    int blocks = (n + 255) / 256;
    if (blocks > 2048) blocks = 2048;
    cvt_bf16_kernel<<<blocks, 256, 0, stream>>>(s, d, n);
  };
  cvt(emb,   emb_bf,   VOCAB1 * En);
  cvt(WQ,    wq_bf,    An * En);
  cvt(WK,    wk_bf,    An * En);
  cvt(Wih_f, wih_f_bf, 4 * Hn * En);
  cvt(Whh_f, whh_f_bf, 4 * Hn * Hn);
  cvt(Wih_b, wih_b_bf, 4 * Hn * En);
  cvt(Whh_b, whh_b_bf, 4 * Hn * Hn);

  attn_kernel<<<Bn * Vn, 256, 0, stream>>>(codes, code_mask, time_vals, decay,
                                           initial, emb_bf, wq_bf, wk_bf, vv_bf);

  xg_kernel<<<dim3((Bn * Vn) / 16, 2), 256, 0, stream>>>(vv_bf, wih_f_bf, wih_b_bf,
                                                         b_f, b_b, xg);

  lstm_kernel<<<2, 256, 0, stream>>>(whh_f_bf, whh_b_bf, xg, visit_mask, out);
}
